// FixTransformerBlock_44959717654746
// MI455X (gfx1250) — compile-verified, matrix operand order checked by reading, not run on hardware
//
#include <hip/hip_runtime.h>
#include <math.h>

typedef unsigned short ushort_t;
typedef unsigned int uint_t;

typedef __bf16 v16bf __attribute__((ext_vector_type(16)));
typedef float  v8f   __attribute__((ext_vector_type(8)));
typedef int    v4i_t __attribute__((ext_vector_type(4)));

#define L_TOK   262144          // B*H*W = 4*256*256
#define C_DIM   96
#define NHEAD   3
#define HD      32
#define HID     384
#define WIN     8
#define NWINTOK 64              // WIN*WIN
#define SCALE_Q 0.17677669529663687f   // 32^-0.5

#if defined(__has_builtin)
#if __has_builtin(__builtin_amdgcn_global_load_async_to_lds_b128)
#define USE_ASYNC_LDS 1
#endif
#endif

#define AS_GLOBAL_V4I(p) ((__attribute__((address_space(1))) v4i_t*)(p))
#define AS_LDS_V4I(p)    ((__attribute__((address_space(3))) v4i_t*)(p))

// ---------- small helpers ----------
__device__ inline ushort_t f2bf(float f) {
    uint_t u = __float_as_uint(f);
    uint_t r = (u + 0x7FFFu + ((u >> 16) & 1u)) >> 16;   // RNE
    return (ushort_t)r;
}
__device__ inline float bf2f(ushort_t h) {
    return __uint_as_float(((uint_t)h) << 16);
}
__device__ inline float gelu_exact(float x) {
    return 0.5f * x * (1.0f + erff(x * 0.70710678118654752f));
}

__device__ inline v8f wmma_bf16(v16bf a, v16bf b, v8f c) {
    // D(16x16,f32) = A(16x32,bf16) * B(32x16,bf16) + C
    return __builtin_amdgcn_wmma_f32_16x16x32_bf16(
        false, a, false, b, (short)0, c, false, false);
}

__device__ inline void wait_async_lds() {
#if defined(USE_ASYNC_LDS)
#if __has_builtin(__builtin_amdgcn_s_wait_asynccnt)
    __builtin_amdgcn_s_wait_asynccnt(0);
#else
    asm volatile("s_wait_asynccnt 0x0" ::: "memory");
#endif
#endif
}

// Load a 16x32 bf16 tile in WMMA A-operand layout from a row-major buffer.
// p points at tile origin; ld = row stride in elements.
// Per ISA: lanes 0-15 hold M=lane, K 0-7 & 16-23; lanes 16-31 hold M=lane-16,
// K 8-15 & 24-31  ->  two 16B loads per lane.
__device__ inline v16bf load_tileA(const ushort_t* p, int ld) {
    int lane = threadIdx.x & 31;
    int half = lane >> 4;
    int m    = lane & 15;
    const ushort_t* row = p + (size_t)m * ld;
    union { uint4 u[2]; v16bf v; } t;
    t.u[0] = *(const uint4*)(row + half * 8);
    t.u[1] = *(const uint4*)(row + 16 + half * 8);
    return t.v;
}

// ---------- weight prep ----------
// dst[n*K + k] = (bf16) src[k*N + n]   (store W^T so B-tiles load like A-tiles)
__global__ void k_transpose_bf16(const float* __restrict__ src,
                                 ushort_t* __restrict__ dst, int K, int N) {
    int idx = blockIdx.x * blockDim.x + threadIdx.x;
    if (idx >= K * N) return;
    int n = idx / K, k = idx % K;
    dst[idx] = f2bf(src[(size_t)k * N + n]);
}

__global__ void k_concat_bias(const float* __restrict__ bq,
                              const float* __restrict__ bkv,
                              float* __restrict__ dst) {
    int i = blockIdx.x * blockDim.x + threadIdx.x;
    if (i < C_DIM) dst[i] = bq[i];
    else if (i < 3 * C_DIM) dst[i] = bkv[i - C_DIM];
}

// ---------- LayerNorm (one wave32 per token, C=96 -> 3 elems/lane) ----------
__global__ __launch_bounds__(256)
void k_layernorm_bf16(const float* __restrict__ x, const float* __restrict__ g,
                      const float* __restrict__ b, ushort_t* __restrict__ out) {
    int tok  = (blockIdx.x * blockDim.x + threadIdx.x) >> 5;
    int lane = threadIdx.x & 31;
    if (tok >= L_TOK) return;
    const float* row = x + (size_t)tok * C_DIM;
    float v0 = row[lane], v1 = row[lane + 32], v2 = row[lane + 64];
    float s = v0 + v1 + v2;
#pragma unroll
    for (int o = 16; o; o >>= 1) s += __shfl_xor(s, o, 32);
    float mean = s * (1.0f / 96.0f);
    float d0 = v0 - mean, d1 = v1 - mean, d2 = v2 - mean;
    float q = d0 * d0 + d1 * d1 + d2 * d2;
#pragma unroll
    for (int o = 16; o; o >>= 1) q += __shfl_xor(q, o, 32);
    float rstd = rsqrtf(q * (1.0f / 96.0f) + 1e-5f);
    ushort_t* orow = out + (size_t)tok * C_DIM;
    orow[lane]      = f2bf(d0 * rstd * g[lane]      + b[lane]);
    orow[lane + 32] = f2bf(d1 * rstd * g[lane + 32] + b[lane + 32]);
    orow[lane + 64] = f2bf(d2 * rstd * g[lane + 64] + b[lane + 64]);
}

// ---------- generic WMMA GEMM:  out = epilogue(A[MxK]bf16 @ Wt[NxK]^T + bias)
// EPI 0: qkv   -> scale cols [0,96) by SCALE_Q, write bf16
// EPI 1: resid -> out.f32 = acc + bias + extra (f32 residual)
// EPI 2: gelu  -> write bf16(gelu(acc + bias))
template <int EPI>
__global__ __launch_bounds__(256)
void k_gemm_bf16(const ushort_t* __restrict__ A, const ushort_t* __restrict__ Wt,
                 const float* __restrict__ bias, const float* __restrict__ extra,
                 void* __restrict__ out, int M, int K, int N) {
    int wv   = threadIdx.x >> 5;
    int lane = threadIdx.x & 31;
    int half = lane >> 4;
    int nn   = lane & 15;
    int m0 = blockIdx.x * 128 + wv * 16;
    int n0 = blockIdx.y * 16;
    if (m0 >= M) return;
    const ushort_t* Ab = A  + (size_t)m0 * K;
    const ushort_t* Bb = Wt + (size_t)n0 * K;
    v8f c = {};
    for (int k = 0; k < K; k += 32) {
        if (k + 32 < K) {   // pull next K-slice while WMMA issues
            __builtin_prefetch(Ab + k + 32 + (size_t)(lane & 15) * K, 0, 3);
            __builtin_prefetch(Bb + k + 32 + (size_t)(lane & 15) * K, 0, 3);
        }
        v16bf a = load_tileA(Ab + k, K);
        v16bf b = load_tileA(Bb + k, K);
        c = wmma_bf16(a, b, c);
    }
    int col = n0 + nn;
    float bs = bias[col];
#pragma unroll
    for (int r = 0; r < 8; ++r) {
        size_t row = (size_t)m0 + r + 8 * half;
        float v = c[r] + bs;
        if constexpr (EPI == 0) {
            if (col < C_DIM) v *= SCALE_Q;
            ((ushort_t*)out)[row * N + col] = f2bf(v);
        } else if constexpr (EPI == 1) {
            ((float*)out)[row * N + col] = v + extra[row * N + col];
        } else {
            ((ushort_t*)out)[row * N + col] = f2bf(gelu_exact(v));
        }
    }
}

// ---------- window attention: one block per (window, head) ----------
__global__ __launch_bounds__(128)
void k_attention(const ushort_t* __restrict__ qkv, const float* __restrict__ rpb,
                 ushort_t* __restrict__ yw) {
    __shared__ __align__(16) ushort_t qs[NWINTOK * HD];   // 64x32
    __shared__ __align__(16) ushort_t ks[NWINTOK * HD];   // 64x32
    __shared__ __align__(16) ushort_t vT[HD * NWINTOK];   // 32x64 (v transposed)
    __shared__ __align__(16) float    biasS[NWINTOK * NWINTOK];
    __shared__ __align__(16) float    scoreS[NWINTOK * NWINTOK];
    __shared__ __align__(16) ushort_t probS[NWINTOK * NWINTOK];

    int bid  = blockIdx.x;
    int head = bid % NHEAD;
    int win  = bid / NHEAD;
    int b    = win >> 10;            // 1024 windows per batch image
    int rem  = win & 1023;
    int wh   = rem >> 5;             // 32 windows per row
    int ww   = rem & 31;
    int tid  = threadIdx.x;

    // --- stage q, k into LDS via async copy; v transposed via VALU path ---
    {
        int n     = tid >> 1;        // token in window 0..63
        int halfc = tid & 1;         // which 16-col half of HD
        int i = n >> 3, j = n & 7;
        size_t token = ((size_t)b << 16) + (size_t)(wh * WIN + i) * 256 + (ww * WIN + j);
        const ushort_t* base = qkv + token * (3 * C_DIM) + head * HD + halfc * 16;
        ushort_t* qdst = qs + n * HD + halfc * 16;
        ushort_t* kdst = ks + n * HD + halfc * 16;
#if defined(USE_ASYNC_LDS)
        __builtin_amdgcn_global_load_async_to_lds_b128(
            AS_GLOBAL_V4I(base), AS_LDS_V4I(qdst), 0, 0);
        __builtin_amdgcn_global_load_async_to_lds_b128(
            AS_GLOBAL_V4I(base + 8), AS_LDS_V4I(qdst + 8), 0, 0);
        __builtin_amdgcn_global_load_async_to_lds_b128(
            AS_GLOBAL_V4I(base + C_DIM), AS_LDS_V4I(kdst), 0, 0);
        __builtin_amdgcn_global_load_async_to_lds_b128(
            AS_GLOBAL_V4I(base + C_DIM + 8), AS_LDS_V4I(kdst + 8), 0, 0);
#else
        *(uint4*)(qdst)     = *(const uint4*)(base);
        *(uint4*)(qdst + 8) = *(const uint4*)(base + 8);
        *(uint4*)(kdst)     = *(const uint4*)(base + C_DIM);
        *(uint4*)(kdst + 8) = *(const uint4*)(base + C_DIM + 8);
#endif
        const ushort_t* vb = base + 2 * C_DIM;
#pragma unroll
        for (int cc = 0; cc < 16; ++cc)
            vT[(halfc * 16 + cc) * NWINTOK + n] = vb[cc];
    }
    // --- relative-position bias table (overlaps with async staging) ---
    for (int e = tid; e < NWINTOK * NWINTOK; e += 128) {
        int n = e >> 6, m = e & 63;
        int idx = ((n >> 3) - (m >> 3) + 7) * 15 + ((n & 7) - (m & 7) + 7);
        biasS[e] = rpb[idx * NHEAD + head];
    }
    wait_async_lds();
    __syncthreads();

    int wv = tid >> 5, lane = tid & 31, half = lane >> 4, nn = lane & 15;
    int n0 = wv * 16;

    // --- scores = (q*SCALE) @ k^T + bias   (K = HD = 32 -> 1 WMMA/tile) ---
    {
        v16bf a = load_tileA(qs + n0 * HD, HD);
        for (int ct = 0; ct < 4; ++ct) {
            v16bf bb = load_tileA(ks + ct * 16 * HD, HD);
            v8f c = {};
            c = wmma_bf16(a, bb, c);
            int col = ct * 16 + nn;
#pragma unroll
            for (int r = 0; r < 8; ++r) {
                int row = n0 + r + 8 * half;
                scoreS[row * NWINTOK + col] = c[r] + biasS[row * NWINTOK + col];
            }
        }
    }
    __syncthreads();

    // --- softmax over rows (one thread per row) ---
    if (tid < NWINTOK) {
        float mx = -3.0e38f;
        for (int m = 0; m < NWINTOK; ++m) mx = fmaxf(mx, scoreS[tid * NWINTOK + m]);
        float s = 0.0f;
        for (int m = 0; m < NWINTOK; ++m) {
            float e = __expf(scoreS[tid * NWINTOK + m] - mx);
            scoreS[tid * NWINTOK + m] = e;
            s += e;
        }
        float inv = 1.0f / s;
        for (int m = 0; m < NWINTOK; ++m)
            probS[tid * NWINTOK + m] = f2bf(scoreS[tid * NWINTOK + m] * inv);
    }
    __syncthreads();

    // --- y = P(64x64) @ v(64x32)   (2 k-steps, 2 col-tiles) ---
    for (int ct = 0; ct < 2; ++ct) {
        v8f c = {};
#pragma unroll
        for (int kk = 0; kk < 2; ++kk) {
            v16bf a  = load_tileA(probS + n0 * NWINTOK + kk * 32, NWINTOK);
            v16bf bb = load_tileA(vT + (ct * 16) * NWINTOK + kk * 32, NWINTOK);
            c = wmma_bf16(a, bb, c);
        }
        int colh = head * HD + ct * 16 + nn;
#pragma unroll
        for (int r = 0; r < 8; ++r) {
            int n = n0 + r + 8 * half;
            int i = n >> 3, j = n & 7;
            size_t token = ((size_t)b << 16) + (size_t)(wh * WIN + i) * 256 + (ww * WIN + j);
            yw[token * C_DIM + colh] = f2bf(c[r]);
        }
    }
}

// ---------- depthwise 3x3 conv + GELU (NHWC bf16, 2 channels/thread) ----------
__global__ __launch_bounds__(256)
void k_dwconv_gelu(const ushort_t* __restrict__ h1, const float* __restrict__ wdw,
                   const float* __restrict__ bdw, ushort_t* __restrict__ h2) {
    size_t gid = (size_t)blockIdx.x * blockDim.x + threadIdx.x;
    if (gid >= (size_t)L_TOK * (HID / 2)) return;
    int cc = (int)(gid % (HID / 2));
    size_t t = gid / (HID / 2);
    int c0 = cc * 2;
    int b = (int)(t >> 16);
    int hw = (int)(t & 65535);
    int i = hw >> 8, j = hw & 255;
    float acc0 = bdw[c0], acc1 = bdw[c0 + 1];
#pragma unroll
    for (int di = -1; di <= 1; ++di) {
        int ii = i + di;
        if ((unsigned)ii >= 256u) continue;
#pragma unroll
        for (int dj = -1; dj <= 1; ++dj) {
            int jj = j + dj;
            if ((unsigned)jj >= 256u) continue;
            size_t tok = ((size_t)b << 16) + ((size_t)ii << 8) + jj;
            uint_t pv = *(const uint_t*)(h1 + tok * HID + c0);
            int wi = (di + 1) * 3 + (dj + 1);
            acc0 += wdw[c0 * 9 + wi]       * bf2f((ushort_t)(pv & 0xFFFF));
            acc1 += wdw[(c0 + 1) * 9 + wi] * bf2f((ushort_t)(pv >> 16));
        }
    }
    uint_t packed = (uint_t)f2bf(gelu_exact(acc0)) | ((uint_t)f2bf(gelu_exact(acc1)) << 16);
    *(uint_t*)(h2 + t * HID + c0) = packed;
}

// ---------- host ----------
extern "C" void kernel_launch(void* const* d_in, const int* in_sizes, int n_in,
                              void* d_out, int out_size, void* d_ws, size_t ws_size,
                              hipStream_t stream) {
    const float* x   = (const float*)d_in[0];
    const float* g1  = (const float*)d_in[1];
    const float* b1  = (const float*)d_in[2];
    const float* wq  = (const float*)d_in[3];
    const float* bq  = (const float*)d_in[4];
    const float* wkv = (const float*)d_in[5];
    const float* bkv = (const float*)d_in[6];
    const float* rpb = (const float*)d_in[7];
    const float* wp  = (const float*)d_in[8];
    const float* bp  = (const float*)d_in[9];
    const float* g2  = (const float*)d_in[10];
    const float* b2  = (const float*)d_in[11];
    const float* w1  = (const float*)d_in[12];
    const float* bl1 = (const float*)d_in[13];
    const float* wdw = (const float*)d_in[14];
    const float* bdw = (const float*)d_in[15];
    const float* w2  = (const float*)d_in[16];
    const float* bl2 = (const float*)d_in[17];
    float* out = (float*)d_out;

    const size_t L = L_TOK;
    char* ws = (char*)d_ws;
    size_t off = 0;
    auto alloc = [&](size_t bytes) {
        size_t o = off;
        off = (off + bytes + 255) & ~(size_t)255;
        return o;
    };
    size_t oWcat = alloc((size_t)3 * C_DIM * C_DIM * 2);   // 288x96 bf16 (W_qkv^T)
    size_t oWp   = alloc((size_t)C_DIM * C_DIM * 2);       // 96x96 bf16
    size_t oW1   = alloc((size_t)HID * C_DIM * 2);         // 384x96 bf16
    size_t oW2   = alloc((size_t)C_DIM * HID * 2);         // 96x384 bf16
    size_t oBias = alloc((size_t)3 * C_DIM * 4);           // 288 f32
    size_t oXn   = alloc(L * C_DIM * 2);                   // xn1, later reused as xn2
    size_t oQkv  = alloc(L * 3 * C_DIM * 2);               // qkv bf16
    size_t oYw   = alloc(L * C_DIM * 2);                   // attn out (contiguous after qkv)
    size_t oX2   = alloc(L * C_DIM * 4);                   // f32 residual stream
    size_t oH2   = alloc(L * HID * 2);                     // conv out bf16
    size_t oH1   = oQkv;                                   // alias: qkv+yw dead by MLP time
    (void)ws_size; (void)in_sizes; (void)n_in; (void)out_size;

    ushort_t* Wcat  = (ushort_t*)(ws + oWcat);
    ushort_t* WpT   = (ushort_t*)(ws + oWp);
    ushort_t* W1T   = (ushort_t*)(ws + oW1);
    ushort_t* W2T   = (ushort_t*)(ws + oW2);
    float*    biasC = (float*)(ws + oBias);
    ushort_t* xn    = (ushort_t*)(ws + oXn);
    ushort_t* qkv   = (ushort_t*)(ws + oQkv);
    ushort_t* yw    = (ushort_t*)(ws + oYw);
    float*    x2    = (float*)(ws + oX2);
    ushort_t* h1    = (ushort_t*)(ws + oH1);
    ushort_t* h2    = (ushort_t*)(ws + oH2);

    // 1) weight prep: W^T in bf16
    k_transpose_bf16<<<(96 * 96 + 255) / 256, 256, 0, stream>>>(wq, Wcat, 96, 96);
    k_transpose_bf16<<<(96 * 192 + 255) / 256, 256, 0, stream>>>(wkv, Wcat + 96 * 96, 96, 192);
    k_transpose_bf16<<<(96 * 96 + 255) / 256, 256, 0, stream>>>(wp, WpT, 96, 96);
    k_transpose_bf16<<<(96 * 384 + 255) / 256, 256, 0, stream>>>(w1, W1T, 96, 384);
    k_transpose_bf16<<<(384 * 96 + 255) / 256, 256, 0, stream>>>(w2, W2T, 384, 96);
    k_concat_bias<<<3, 128, 0, stream>>>(bq, bkv, biasC);

    // 2) LN1
    k_layernorm_bf16<<<L / 8, 256, 0, stream>>>(x, g1, b1, xn);

    // 3) fused QKV GEMM (SCALE folded into q)
    {
        dim3 g((unsigned)(L / 128), (3 * C_DIM) / 16);
        k_gemm_bf16<0><<<g, 256, 0, stream>>>(xn, Wcat, biasC, nullptr, qkv,
                                              (int)L, C_DIM, 3 * C_DIM);
    }
    // 4) window attention
    k_attention<<<4096 * NHEAD, 128, 0, stream>>>(qkv, rpb, yw);

    // 5) output projection + residual -> x2 (f32)
    {
        dim3 g((unsigned)(L / 128), C_DIM / 16);
        k_gemm_bf16<1><<<g, 256, 0, stream>>>(yw, WpT, bp, x, x2, (int)L, C_DIM, C_DIM);
    }
    // 6) LN2
    k_layernorm_bf16<<<L / 8, 256, 0, stream>>>(x2, g2, b2, xn);

    // 7) MLP up + GELU -> h1 (bf16)  [h1 aliases qkv+yw, both dead now]
    {
        dim3 g((unsigned)(L / 128), HID / 16);
        k_gemm_bf16<2><<<g, 256, 0, stream>>>(xn, W1T, bl1, nullptr, h1, (int)L, C_DIM, HID);
    }
    // 8) depthwise 3x3 conv + GELU -> h2
    {
        size_t total = L * (HID / 2);
        k_dwconv_gelu<<<(unsigned)((total + 255) / 256), 256, 0, stream>>>(h1, wdw, bdw, h2);
    }
    // 9) MLP down + residual -> out (f32)
    {
        dim3 g((unsigned)(L / 128), C_DIM / 16);
        k_gemm_bf16<1><<<g, 256, 0, stream>>>(h2, W2T, bl2, x2, out, (int)L, HID, C_DIM);
    }
}
